// Regularizedpooling_81836306858645
// MI455X (gfx1250) — matrix operand order repaired and checked by reading, not compiled, not run on hardware
//
#include <hip/hip_runtime.h>

typedef __attribute__((ext_vector_type(16))) _Float16 v16h;
typedef __attribute__((ext_vector_type(8)))  _Float16 v8h;
typedef __attribute__((ext_vector_type(8)))  float    v8f;

#define BATCH 32
#define HW    256
#define HO    128
#define KTOT  576   // 9 taps * 64 cin
#define KSTEPS 18   // 576 / 32

// ---------------------------------------------------------------------------
// Repack W2 (OIHW fp32) -> f16 [o][k] with k = (dy*3+dx)*64 + cin
// ---------------------------------------------------------------------------
__global__ void k_w2_cvt(const float* __restrict__ W2, _Float16* __restrict__ w2h) {
    int t = blockIdx.x * 256 + threadIdx.x;
    if (t >= 64 * KTOT) return;
    int o  = t / KTOT;
    int k  = t - o * KTOT;
    int tap = k >> 6, ci = k & 63;
    int dy = tap / 3, dx = tap - dy * 3;
    w2h[t] = (_Float16)W2[((o * 64 + ci) * 3 + dy) * 3 + dx];
}

// ---------------------------------------------------------------------------
// conv1: 1 -> 64 ch, 3x3 SAME, bias + ReLU, output NHWC f16
// ---------------------------------------------------------------------------
__global__ void k_conv1(const float* __restrict__ x, const float* __restrict__ W1,
                        const float* __restrict__ b1, _Float16* __restrict__ f1) {
    int t   = blockIdx.x * 256 + threadIdx.x;
    int pix = t >> 4;
    int cq  = (t & 15) << 2;
    int b   = pix >> 16;
    int rem = pix & 0xFFFF;
    int y   = rem >> 8, xx = rem & 255;
    const float* xb = x + (size_t)b * (HW * HW);
    float in[9];
    #pragma unroll
    for (int dy = 0; dy < 3; ++dy)
        #pragma unroll
        for (int dx = 0; dx < 3; ++dx) {
            int yy = y + dy - 1, xc = xx + dx - 1;
            bool ok = ((unsigned)yy < HW) && ((unsigned)xc < HW);
            in[dy * 3 + dx] = ok ? xb[yy * HW + xc] : 0.0f;
        }
    union { _Float16 h[4]; uint2 u; } o;
    #pragma unroll
    for (int j = 0; j < 4; ++j) {
        int c = cq + j;
        float s = b1[c];
        #pragma unroll
        for (int tp = 0; tp < 9; ++tp) s = fmaf(in[tp], W1[c * 9 + tp], s);
        o.h[j] = (_Float16)fmaxf(s, 0.0f);
    }
    *reinterpret_cast<uint2*>(f1 + (size_t)pix * 64 + cq) = o.u;
}

// ---------------------------------------------------------------------------
// conv2: 64 -> 64 ch 3x3 SAME via implicit-GEMM WMMA, fused bias+ReLU.
// B tile staged with GLOBAL_LOAD_ASYNC_TO_LDS_B64 (ASYNCcnt path, no VGPR
// round-trip); main loop 2-deep software pipelined so next-step LDS/global
// fragment loads fly behind the current step's 4 WMMAs.
// block = (b, h, 128-px half-row); 8 waves = 4 Mtiles x 2 Nhalves
// ---------------------------------------------------------------------------
__global__ void k_conv2(const _Float16* __restrict__ f1, const _Float16* __restrict__ w2h,
                        const float* __restrict__ b2, float* __restrict__ patches) {
    __shared__ __align__(32) _Float16 Bsm[3 * 130 * 64];   // 49,920 B

    int bid = blockIdx.x;
    int b   = bid >> 9;
    int r   = bid & 511;
    int h   = r >> 1;
    int x0  = (r & 1) << 7;
    int tid = threadIdx.x;

    // --- stage B tile: rows h-1..h+1, px x0-1..x0+128, 64 ch (f16)
    // in-range units: async global->LDS (8B per lane); OOB halo: zero via DS
    for (int u = tid; u < 3 * 130 * 16; u += 256) {
        int dy = u / (130 * 16);
        int rm = u - dy * (130 * 16);
        int p  = rm >> 4;
        int cq = (rm & 15) << 2;
        int yy = h + dy - 1;
        int xc = x0 + p - 1;
        _Float16* lp = &Bsm[(dy * 130 + p) * 64 + cq];
        if (((unsigned)yy < HW) && ((unsigned)xc < HW)) {
            unsigned laddr = (unsigned)(size_t)lp;   // LDS byte offset (low 32 bits)
            unsigned long long ga =
                (unsigned long long)(size_t)(f1 + (((size_t)b * HW + yy) * HW + xc) * 64 + cq);
            asm volatile("global_load_async_to_lds_b64 %0, %1, off"
                         :: "v"(laddr), "v"(ga) : "memory");
        } else {
            *reinterpret_cast<uint2*>(lp) = make_uint2(0u, 0u);
        }
    }
    asm volatile("s_wait_asynccnt 0x0" ::: "memory");
    __syncthreads();

    int wave  = tid >> 5;
    int lane  = tid & 31;
    int mtile = wave & 3;
    int nhalf = wave >> 2;
    int nl    = lane & 15;   // A-row (M) and B-col (N) index per ISA layout
    int hi    = lane >> 4;
    int nbase = nhalf * 64 + nl;

    v8f acc0 = {}, acc1 = {}, acc2 = {}, acc3 = {};
    const _Float16* arow = w2h + (size_t)(mtile * 16 + nl) * KTOT;

    union AF { v16h v; v8h p[2]; };
    AF aEv, aOd;
    v16h be0, be1, be2, be3, bo0, bo1, bo2, bo3;

    // A fragment (16x32 f16): one contiguous 16-half run at hi*16
    #define LOAD_A(KK, A_) do {                                            \
        (A_).p[0] = *reinterpret_cast<const v8h*>(arow + (KK)*32 + hi*16);  \
        (A_).p[1] = *reinterpret_cast<const v8h*>(arow + (KK)*32 + hi*16 + 8); } while (0)
    // B fragment (32x16 f16): one contiguous 16-half run per lane
    #define LOAD_B(KK, B0_, B1_, B2_, B3_) do {                            \
        int tap_ = (KK) >> 1; int dy_ = tap_ / 3; int dx_ = tap_ - dy_*3;  \
        int off_ = dy_*(130*64) + (((KK)&1)<<5) + hi*16 + (nbase + dx_)*64;\
        B0_ = *reinterpret_cast<const v16h*>(&Bsm[off_]);                  \
        B1_ = *reinterpret_cast<const v16h*>(&Bsm[off_ + 16*64]);          \
        B2_ = *reinterpret_cast<const v16h*>(&Bsm[off_ + 32*64]);          \
        B3_ = *reinterpret_cast<const v16h*>(&Bsm[off_ + 48*64]); } while (0)
    #define WMMA4(A_, B0_, B1_, B2_, B3_) do {                                             \
        acc0 = __builtin_amdgcn_wmma_f32_16x16x32_f16(false, (A_).v, false, B0_, (short)0, \
                                                      acc0, false, false);                 \
        acc1 = __builtin_amdgcn_wmma_f32_16x16x32_f16(false, (A_).v, false, B1_, (short)0, \
                                                      acc1, false, false);                 \
        acc2 = __builtin_amdgcn_wmma_f32_16x16x32_f16(false, (A_).v, false, B2_, (short)0, \
                                                      acc2, false, false);                 \
        acc3 = __builtin_amdgcn_wmma_f32_16x16x32_f16(false, (A_).v, false, B3_, (short)0, \
                                                      acc3, false, false); } while (0)

    LOAD_A(0, aEv);
    LOAD_B(0, be0, be1, be2, be3);
    #pragma unroll
    for (int kk = 0; kk < KSTEPS; kk += 2) {
        LOAD_A(kk + 1, aOd);
        LOAD_B(kk + 1, bo0, bo1, bo2, bo3);
        WMMA4(aEv, be0, be1, be2, be3);
        if (kk + 2 < KSTEPS) {
            LOAD_A(kk + 2, aEv);
            LOAD_B(kk + 2, be0, be1, be2, be3);
        }
        WMMA4(aOd, bo0, bo1, bo2, bo3);
    }
    #undef LOAD_A
    #undef LOAD_B
    #undef WMMA4

    // --- epilogue: bias + ReLU, store in pooled-patch layout [B][C][128][128][4]
    int ho   = h >> 1;
    int kRow = (h & 1) << 1;
    float bias[8];
    #pragma unroll
    for (int rr = 0; rr < 8; ++rr) bias[rr] = b2[mtile * 16 + hi * 8 + rr];
    v8f* accs[4] = { &acc0, &acc1, &acc2, &acc3 };
    #pragma unroll
    for (int nt = 0; nt < 4; ++nt) {
        int xg = x0 + nhalf * 64 + nt * 16 + nl;
        int wo = xg >> 1;
        int kp = kRow + (xg & 1);
        v8f a = *accs[nt];
        #pragma unroll
        for (int rr = 0; rr < 8; ++rr) {
            int c = mtile * 16 + hi * 8 + rr;   // D layout: M = r + 8*hi
            float vv = fmaxf(a[rr] + bias[rr], 0.0f);
            patches[((((size_t)b * 64 + c) * HO + ho) * HO + wo) * 4 + kp] = vv;
        }
    }
}

// ---------------------------------------------------------------------------
// pool: 2x2 max + first-max argmax; d = care ? argmax : 0 (1 byte)
// ---------------------------------------------------------------------------
__global__ void k_pool(const float* __restrict__ patches, unsigned char* __restrict__ dbuf) {
    int t = blockIdx.x * 256 + threadIdx.x;
    const float4 v = *reinterpret_cast<const float4*>(patches + (size_t)t * 4);
    float m = v.x; int p = 0;
    if (v.y > m) { m = v.y; p = 1; }
    if (v.z > m) { m = v.z; p = 2; }
    if (v.w > m) { m = v.w; p = 3; }
    dbuf[t] = (m != 0.0f) ? (unsigned char)p : (unsigned char)0;
}

// ---------------------------------------------------------------------------
// smooth + gather: 3x3 zero-padded box mean of displacements;
// round-half-even over s/9 reduces to (s >= 5); gather smoothed slot
// ---------------------------------------------------------------------------
__global__ void k_smooth(const float* __restrict__ patches,
                         const unsigned char* __restrict__ dbuf,
                         float* __restrict__ out) {
    int t  = blockIdx.x * 256 + threadIdx.x;
    int wo = t & 127;
    int ho = (t >> 7) & 127;
    int bc = t >> 14;
    int sx = 0, sy = 0;
    #pragma unroll
    for (int i = -1; i <= 1; ++i)
        #pragma unroll
        for (int j = -1; j <= 1; ++j) {
            int hh = ho + i, ww = wo + j;
            if (((unsigned)hh < 128u) && ((unsigned)ww < 128u)) {
                int d = dbuf[((size_t)bc << 14) + (hh << 7) + ww];
                sx += d & 1;
                sy += d >> 1;
            }
        }
    int ax = (sx >= 5) ? 1 : 0;
    int ay = (sy >= 5) ? 1 : 0;
    out[t] = patches[(size_t)t * 4 + (ay * 2 + ax)];
}

// ---------------------------------------------------------------------------
extern "C" void kernel_launch(void* const* d_in, const int* in_sizes, int n_in,
                              void* d_out, int out_size, void* d_ws, size_t ws_size,
                              hipStream_t stream) {
    const float* x  = (const float*)d_in[0];
    const float* W1 = (const float*)d_in[1];
    const float* b1 = (const float*)d_in[2];
    const float* W2 = (const float*)d_in[3];
    const float* b2 = (const float*)d_in[4];
    char* ws = (char*)d_ws;

    _Float16*      f1      = (_Float16*)(ws);                      // 268,435,456 B
    float*         patches = (float*)(ws + 268435456ull);          // 536,870,912 B
    unsigned char* dbuf    = (unsigned char*)(ws + 805306368ull);  //  33,554,432 B
    _Float16*      w2h     = (_Float16*)(ws + 838860800ull);       //      73,728 B
    float*         out     = (float*)d_out;

    k_w2_cvt <<<144,    256, 0, stream>>>(W2, w2h);
    k_conv1  <<<131072, 256, 0, stream>>>(x, W1, b1, f1);
    k_conv2  <<<16384,  256, 0, stream>>>(f1, w2h, b2, patches);
    k_pool   <<<131072, 256, 0, stream>>>(patches, dbuf);
    k_smooth <<<131072, 256, 0, stream>>>(patches, dbuf, out);
}
